// Mamba2Block_68195490726183
// MI455X (gfx1250) — compile-verified, matrix operand order checked by reading, not run on hardware
//
#include <hip/hip_runtime.h>
#include <hip/hip_bf16.h>

// ---------------- model dims ----------------
#define D_MODEL 1024
#define D_STATE 128
#define D_INNER 2048
#define HEADS   1024
#define HEAD_DIM 2
#define KERNEL_W 4
#define PROJ_OUT 5376   // 2*D_INNER + 2*D_STATE + D_MODEL
// zxbcdt column offsets
#define OFF_X  0
#define OFF_Z  2048
#define OFF_B  4096
#define OFF_C  4224
#define OFF_DT 4352

typedef _Float16 f16;
typedef __attribute__((ext_vector_type(16))) _Float16 v16h;
typedef __attribute__((ext_vector_type(8)))  float    v8f;
typedef __attribute__((ext_vector_type(4)))  unsigned int v4u;
typedef __attribute__((ext_vector_type(8)))  int v8i;
typedef __attribute__((ext_vector_type(4)))  int v4i;

#ifdef __has_builtin
#  if __has_builtin(__builtin_amdgcn_tensor_load_to_lds)
#    define USE_TDM 1
#  endif
#endif
#ifndef USE_TDM
#  define USE_TDM 0
#endif

#if USE_TDM
// Issue one TDM 2D tile load: tile_rows x tile_cols elements (2B each),
// row-major dense into LDS at lds_addr; global row stride = stride_elems.
// D# layout per CDNA5 ISA.
__device__ __forceinline__ void tdm_load_tile_f16(unsigned int lds_addr,
                                                  const void* gptr,
                                                  unsigned int tile_rows,
                                                  unsigned int tile_cols,
                                                  unsigned int stride_elems,
                                                  unsigned int dim0_len,
                                                  unsigned int dim1_len)
{
    const unsigned long long ga = (unsigned long long)(uintptr_t)gptr;
    v4u g0;
    g0[0] = 1u;                                        // count=1 (valid), user mode
    g0[1] = lds_addr;                                  // lds_addr[31:0]
    g0[2] = (unsigned int)(ga & 0xFFFFFFFFu);          // global_addr[31:0]
    g0[3] = (unsigned int)((ga >> 32) & 0x1FFFFFFu)    // global_addr[56:32]
          | (2u << 30);                                // type = 2 ("image")
    v8i g1;
    g1[0] = (int)(1u << 16);                           // wg_mask=0, data_size=1 (2B)
    g1[1] = (int)((dim0_len & 0xFFFFu) << 16);         // tensor_dim0[15:0]
    g1[2] = (int)(((dim0_len >> 16) & 0xFFFFu)         // tensor_dim0[31:16]
          | ((dim1_len & 0xFFFFu) << 16));             // tensor_dim1[15:0]
    g1[3] = (int)(((dim1_len >> 16) & 0xFFFFu)         // tensor_dim1[31:16]
          | ((tile_cols & 0xFFFFu) << 16));            // tile_dim0
    g1[4] = (int)(tile_rows & 0xFFFFu);                // tile_dim1 ; tile_dim2=0 (2D)
    g1[5] = (int)stride_elems;                         // tensor_dim0_stride[31:0]
    g1[6] = 0;                                         // stride[47:32]=0; dim1_stride lo=0
    g1[7] = 0;                                         // dim1_stride hi=0
    v4i g2 = {0, 0, 0, 0};
    v4i g3 = {0, 0, 0, 0};
#if defined(__clang_major__) && (__clang_major__ >= 23)
    v8i g4 = {0, 0, 0, 0, 0, 0, 0, 0};
    __builtin_amdgcn_tensor_load_to_lds(g0, g1, g2, g3, g4, 0);
#else
    __builtin_amdgcn_tensor_load_to_lds(g0, g1, g2, g3, 0);
#endif
}
#endif

// ---------------- LayerNorm -> f16 ----------------
__global__ __launch_bounds__(256)
void ln_f16_kernel(const float* __restrict__ u, const float* __restrict__ g,
                   const float* __restrict__ b, f16* __restrict__ out, int D)
{
    __shared__ float red[256];
    const int row = blockIdx.x;
    const float* x = u + (size_t)row * D;
    float s = 0.f;
    for (int c = threadIdx.x; c < D; c += 256) s += x[c];
    red[threadIdx.x] = s; __syncthreads();
    for (int o = 128; o > 0; o >>= 1) {
        if (threadIdx.x < o) red[threadIdx.x] += red[threadIdx.x + o];
        __syncthreads();
    }
    const float mu = red[0] / (float)D;
    __syncthreads();
    float v = 0.f;
    for (int c = threadIdx.x; c < D; c += 256) { float d = x[c] - mu; v += d * d; }
    red[threadIdx.x] = v; __syncthreads();
    for (int o = 128; o > 0; o >>= 1) {
        if (threadIdx.x < o) red[threadIdx.x] += red[threadIdx.x + o];
        __syncthreads();
    }
    const float rstd = rsqrtf(red[0] / (float)D + 1e-5f);
    for (int c = threadIdx.x; c < D; c += 256)
        out[(size_t)row * D + c] = (f16)(((x[c] - mu) * rstd) * g[c] + b[c]);
}

// ---------------- transpose f32(RxC) -> f16(CxR) ----------------
__global__ __launch_bounds__(256)
void transpose_cast_kernel(const float* __restrict__ in, f16* __restrict__ out,
                           int R, int C)
{
    __shared__ float tile[32][33];
    const int c0 = blockIdx.x * 32, r0 = blockIdx.y * 32;
    const int tx = threadIdx.x & 31, ty = threadIdx.x >> 5;   // 32x8 threads
    #pragma unroll
    for (int i = 0; i < 32; i += 8) {
        int r = r0 + ty + i, c = c0 + tx;
        tile[ty + i][tx] = (r < R && c < C) ? in[(size_t)r * C + c] : 0.f;
    }
    __syncthreads();
    #pragma unroll
    for (int i = 0; i < 32; i += 8) {
        int r = c0 + ty + i, c = r0 + tx;         // out is C x R
        if (r < C && c < R) out[(size_t)r * R + c] = (f16)tile[tx][ty + i];
    }
}

// ---------------- WMMA GEMM: C = A(MxK,f16) x Bt(NxK,f16)^T (+resid) ----------------
// TDM double-buffered LDS staging: wave 0 issues tensor_load_to_lds for the
// next K-tile while all 8 waves run WMMAs on the current one; completion via
// s_wait_tensorcnt + workgroup barrier.
#define BM 64
#define BN 128
#define BK 32

__global__ __launch_bounds__(256)
void gemm_wmma_f16f32(const f16* __restrict__ A, const f16* __restrict__ Bt,
                      float* __restrict__ C, const float* __restrict__ resid,
                      int M, int N, int K)
{
    __shared__ f16 As[2][BM * BK];   // 2 x 4 KB
    __shared__ f16 Bs[2][BN * BK];   // 2 x 8 KB
    const int tid  = threadIdx.x;
    const int lane = tid & 31;
    const int wave = tid >> 5;        // 0..7
    const int wm   = wave >> 2;       // 0..1 -> 32-row strip
    const int wn   = wave & 3;        // 0..3 -> 32-col strip
    const int m0 = blockIdx.y * BM;
    const int n0 = blockIdx.x * BN;

    const int lr = lane & 15;         // row (A) / col (B) within 16x16 tile
    const int kb = (lane >> 4) * 8;   // K sub-chunk base per ISA A-frag layout

    v8f acc[2][2] = {};

#if USE_TDM
    // ---- prologue: DMA first K-tile into buffer 0 ----
    if (wave == 0) {
        tdm_load_tile_f16((unsigned int)(uintptr_t)&As[0][0], &A[(size_t)m0 * K],
                          BM, BK, (unsigned int)K, (unsigned int)K, (unsigned int)(M - m0));
        tdm_load_tile_f16((unsigned int)(uintptr_t)&Bs[0][0], &Bt[(size_t)n0 * K],
                          BN, BK, (unsigned int)K, (unsigned int)K, (unsigned int)(N - n0));
        __builtin_amdgcn_s_wait_tensorcnt(0);
    }
    __syncthreads();
#else
    // ---- fallback: cooperative copy of first tile ----
    {
        const int row = tid >> 2, col = (tid & 3) * 8;
        *(uint4*)&As[0][row * BK + col] = *(const uint4*)&A[(size_t)(m0 + row) * K + col];
        #pragma unroll
        for (int i = 0; i < 2; ++i) {
            const int chunk = tid + i * 256;
            const int br = chunk >> 2, bc = (chunk & 3) * 8;
            *(uint4*)&Bs[0][br * BK + bc] = *(const uint4*)&Bt[(size_t)(n0 + br) * K + bc];
        }
    }
    __syncthreads();
#endif

    int buf = 0;
    for (int k0 = 0; k0 < K; k0 += BK) {
        const int nbuf = buf ^ 1;
        const bool have_next = (k0 + BK) < K;

        // ---- kick off DMA / copy of the NEXT tile into the other buffer ----
#if USE_TDM
        if (wave == 0 && have_next) {
            tdm_load_tile_f16((unsigned int)(uintptr_t)&As[nbuf][0],
                              &A[(size_t)m0 * K + k0 + BK],
                              BM, BK, (unsigned int)K,
                              (unsigned int)(K - k0 - BK), (unsigned int)(M - m0));
            tdm_load_tile_f16((unsigned int)(uintptr_t)&Bs[nbuf][0],
                              &Bt[(size_t)n0 * K + k0 + BK],
                              BN, BK, (unsigned int)K,
                              (unsigned int)(K - k0 - BK), (unsigned int)(N - n0));
        }
#else
        if (have_next) {
            const int row = tid >> 2, col = (tid & 3) * 8;
            *(uint4*)&As[nbuf][row * BK + col] =
                *(const uint4*)&A[(size_t)(m0 + row) * K + k0 + BK + col];
            #pragma unroll
            for (int i = 0; i < 2; ++i) {
                const int chunk = tid + i * 256;
                const int br = chunk >> 2, bc = (chunk & 3) * 8;
                *(uint4*)&Bs[nbuf][br * BK + bc] =
                    *(const uint4*)&Bt[(size_t)(n0 + br) * K + k0 + BK + bc];
            }
        }
#endif

        // ---- compute on the CURRENT buffer ----
        v16h afrag[2], bfrag[2];
        #pragma unroll
        for (int i = 0; i < 2; ++i) {
            const f16* ap = &As[buf][(wm * 32 + i * 16 + lr) * BK + kb];
            ((uint4*)&afrag[i])[0] = *(const uint4*)(ap);
            ((uint4*)&afrag[i])[1] = *(const uint4*)(ap + 16);
            const f16* bp = &Bs[buf][(wn * 32 + i * 16 + lr) * BK + kb];
            ((uint4*)&bfrag[i])[0] = *(const uint4*)(bp);
            ((uint4*)&bfrag[i])[1] = *(const uint4*)(bp + 16);
        }
        #pragma unroll
        for (int mi = 0; mi < 2; ++mi)
            #pragma unroll
            for (int ni = 0; ni < 2; ++ni)
                acc[mi][ni] = __builtin_amdgcn_wmma_f32_16x16x32_f16(
                    false, afrag[mi], false, bfrag[ni],
                    (short)0, acc[mi][ni], false, false);

        // ---- retire DMA, publish next buffer to all waves ----
#if USE_TDM
        if (wave == 0) __builtin_amdgcn_s_wait_tensorcnt(0);
#endif
        __syncthreads();
        buf = nbuf;
    }

    // epilogue: C/D frag layout — lane<16: M=tile+r, lane>=16: M=tile+8+r; N=lr
    #pragma unroll
    for (int mi = 0; mi < 2; ++mi)
        #pragma unroll
        for (int ni = 0; ni < 2; ++ni) {
            const int rbase = m0 + wm * 32 + mi * 16 + (lane >> 4) * 8;
            const int col   = n0 + wn * 32 + ni * 16 + lr;
            #pragma unroll
            for (int r = 0; r < 8; ++r) {
                const size_t idx = (size_t)(rbase + r) * N + col;
                float v = acc[mi][ni][r];
                if (resid) v += resid[idx];
                C[idx] = v;
            }
        }
}

// ---------------- causal depthwise conv (K=4) + SiLU ----------------
__global__ __launch_bounds__(256)
void conv_silu_kernel(const float* __restrict__ zxbcdt, const float* __restrict__ w,
                      const float* __restrict__ bias, float* __restrict__ xc,
                      int L, size_t total)
{
    const size_t idx = (size_t)blockIdx.x * 256 + threadIdx.x;  // over B*L*D_INNER
    if (idx >= total) return;
    const int c = (int)(idx % D_INNER);
    const size_t row = idx / D_INNER;       // b*L + t
    const int t = (int)(row % L);
    float acc = bias[c];
    #pragma unroll
    for (int j = 0; j < KERNEL_W; ++j) {
        const int tt = t - (KERNEL_W - 1) + j;
        if (tt >= 0)
            acc += w[c * KERNEL_W + j] *
                   zxbcdt[(row - (KERNEL_W - 1) + j) * PROJ_OUT + OFF_X + c];
    }
    xc[idx] = acc / (1.f + expf(-acc));     // SiLU
}

// ---------------- SSM scan: one wave per (b,h); fused SiLU(z) gate ----------------
__global__ __launch_bounds__(256)
void ssm_scan_kernel(const float* __restrict__ zxbcdt, const float* __restrict__ xc,
                     const float* __restrict__ A_log, f16* __restrict__ g, int L)
{
    const int lane = threadIdx.x & 31;
    const int wave = threadIdx.x >> 5;
    const int gid  = blockIdx.x * 8 + wave;     // 0 .. B*HEADS-1
    const int h    = gid & (HEADS - 1);
    const int b    = gid >> 10;                 // HEADS == 1024
    const float Ah = -expf(A_log[h]);

    float h0[4] = {0.f, 0.f, 0.f, 0.f};
    float h1[4] = {0.f, 0.f, 0.f, 0.f};
    const int n0 = lane * 4;                    // each lane owns 4 states x 2 p
    const size_t rowBase = (size_t)b * L;

    for (int t = 0; t < L; ++t) {
        const float* zrow = zxbcdt + (rowBase + t) * PROJ_OUT;
        // prefetch next timestep's row segments (latency-bound serial loop)
        if (t + 1 < L) {
            __builtin_prefetch(zrow + PROJ_OUT + OFF_B + n0, 0, 1);
            __builtin_prefetch(zrow + PROJ_OUT + OFF_DT + h, 0, 1);
        }
        const float dtraw = zrow[OFF_DT + h];
        const float dt  = (dtraw > 20.f) ? dtraw : log1pf(expf(dtraw)); // softplus
        const float dec = expf(dt * Ah);
        const float* xrow = xc + (rowBase + t) * D_INNER + 2 * h;
        const float sx0 = dt * xrow[0];
        const float sx1 = dt * xrow[1];
        const float4 Bv = *(const float4*)(zrow + OFF_B + n0);
        const float4 Cv = *(const float4*)(zrow + OFF_C + n0);
        const float bb[4] = {Bv.x, Bv.y, Bv.z, Bv.w};
        const float cc[4] = {Cv.x, Cv.y, Cv.z, Cv.w};
        float y0 = 0.f, y1 = 0.f;
        #pragma unroll
        for (int j = 0; j < 4; ++j) {
            h0[j] = dec * h0[j] + sx0 * bb[j];  y0 += h0[j] * cc[j];
            h1[j] = dec * h1[j] + sx1 * bb[j];  y1 += h1[j] * cc[j];
        }
        #pragma unroll
        for (int o = 16; o > 0; o >>= 1) {      // wave32 cross-lane reduction
            y0 += __shfl_xor(y0, o, 32);
            y1 += __shfl_xor(y1, o, 32);
        }
        if (lane == 0) {
            const float z0 = zrow[OFF_Z + 2 * h];
            const float z1 = zrow[OFF_Z + 2 * h + 1];
            f16* go = g + (rowBase + t) * D_INNER + 2 * h;
            go[0] = (f16)(y0 * (z0 / (1.f + expf(-z0))));
            go[1] = (f16)(y1 * (z1 / (1.f + expf(-z1))));
        }
    }
}

// ---------------- host launch ----------------
extern "C" void kernel_launch(void* const* d_in, const int* in_sizes, int n_in,
                              void* d_out, int out_size, void* d_ws, size_t ws_size,
                              hipStream_t stream) {
    const float* u      = (const float*)d_in[0];
    const float* W_in   = (const float*)d_in[1];
    const float* conv_w = (const float*)d_in[2];
    const float* conv_b = (const float*)d_in[3];
    const float* W_out  = (const float*)d_in[4];
    const float* ln_g   = (const float*)d_in[5];
    const float* ln_b   = (const float*)d_in[6];
    const float* A_log  = (const float*)d_in[7];
    float* out = (float*)d_out;

    const int Bb = 2, L = 2048, BL = Bb * L;   // 4096 rows

    // workspace carve-up (~162 MB, all 16B-aligned)
    char* ws = (char*)d_ws;
    f16*   Xh     = (f16*)ws;   ws += (size_t)BL * D_MODEL * 2;        //  8 MB
    f16*   WinT   = (f16*)ws;   ws += (size_t)PROJ_OUT * D_MODEL * 2;  // 11 MB
    f16*   WoutT  = (f16*)ws;   ws += (size_t)D_MODEL * D_INNER * 2;   //  4 MB
    float* zxbcdt = (float*)ws; ws += (size_t)BL * PROJ_OUT * 4;       // 88 MB
    float* xconv  = (float*)ws; ws += (size_t)BL * D_INNER * 4;        // 32 MB
    f16*   gbuf   = (f16*)ws;   ws += (size_t)BL * D_INNER * 2;        // 16 MB

    // 1) LayerNorm -> f16 activations
    ln_f16_kernel<<<BL, 256, 0, stream>>>(u, ln_g, ln_b, Xh, D_MODEL);

    // 2) transpose+cast weights to f16 (N x K) so WMMA B-frags are contiguous in K
    transpose_cast_kernel<<<dim3((PROJ_OUT + 31) / 32, (D_MODEL + 31) / 32), 256, 0, stream>>>(
        W_in, WinT, D_MODEL, PROJ_OUT);
    transpose_cast_kernel<<<dim3((D_MODEL + 31) / 32, (D_INNER + 31) / 32), 256, 0, stream>>>(
        W_out, WoutT, D_INNER, D_MODEL);

    // 3) GEMM1: zxbcdt = Xh @ W_in   (4096x1024 @ 1024x5376)
    gemm_wmma_f16f32<<<dim3(PROJ_OUT / BN, BL / BM), 256, 0, stream>>>(
        Xh, WinT, zxbcdt, nullptr, BL, PROJ_OUT, D_MODEL);

    // 4) depthwise causal conv + SiLU
    const size_t convTotal = (size_t)BL * D_INNER;
    conv_silu_kernel<<<(int)((convTotal + 255) / 256), 256, 0, stream>>>(
        zxbcdt, conv_w, conv_b, xconv, L, convTotal);

    // 5) SSM scan, one wave per (b,h); writes gated f16
    ssm_scan_kernel<<<(Bb * HEADS) / 8, 256, 0, stream>>>(
        zxbcdt, xconv, A_log, gbuf, L);

    // 6) GEMM2 + residual: out = u + g @ W_out  (4096x2048 @ 2048x1024)
    gemm_wmma_f16f32<<<dim3(D_MODEL / BN, BL / BM), 256, 0, stream>>>(
        gbuf, WoutT, out, u, BL, D_MODEL, D_INNER);
}
